// PhysicalLayer_64914135712062
// MI455X (gfx1250) — matrix operand order repaired and checked by reading, not compiled, not run on hardware
//
#include <hip/hip_runtime.h>
#include <hip/hip_bf16.h>
#include <math.h>

typedef __attribute__((ext_vector_type(16))) _Float16 v16h;
typedef __attribute__((ext_vector_type(8)))  _Float16 v8h;
typedef __attribute__((ext_vector_type(8)))  float    v8f;

#define D_PI 3.14159265358979323846

#define NN   500        // physical grid
#define NP   512        // padded grid
#define RRAD 15
#define PP   31         // 2R+1
#define NE   64         // emitters / z planes
#define NB   8          // batch
#define OUTW 198
#define CANV 200

// RI*WL/(N*PX) = 1.0*5.32e-7/(500*1e-6)
#define FREQ_SC 1.064e-3f
// K = 2*pi*RI/WL
#define KWAVE ((float)(2.0 * D_PI / 5.32e-7))

struct HL { v16h h; v16h l; };

__device__ __forceinline__ v8f wmma_acc(v16h a, v16h b, v8f c) {
  return __builtin_amdgcn_wmma_f32_16x16x32_f16(false, a, false, b, (short)0, c,
                                                false, false);
}

// 3-term f16 split product: near-f32 precision (a = ah+al, b = bh+bl)
__device__ __forceinline__ v8f mac3(const HL& a, const HL& b, v8f c) {
  c = wmma_acc(a.h, b.h, c);
  c = wmma_acc(a.h, b.l, c);
  c = wmma_acc(a.l, b.h, c);
  return c;
}

__device__ __forceinline__ void split2(float x, _Float16& h, _Float16& l) {
  h = (_Float16)x;
  l = (_Float16)(x - (float)h);
}

__device__ __forceinline__ void splitw(float x, v16h& h, v16h& l, int e) {
  _Float16 hv = (_Float16)x;
  h[e] = hv;
  l[e] = (_Float16)(x - (float)hv);
}

// two contiguous 8-half groups -> one 16-half WMMA operand (b128 loads)
__device__ __forceinline__ v16h ld16(const _Float16* __restrict__ p0,
                                     const _Float16* __restrict__ p1) {
  v8h a = *(const v8h*)p0;
  v8h b = *(const v8h*)p1;
  return __builtin_shufflevector(a, b, 0, 1, 2, 3, 4, 5, 6, 7,
                                 8, 9, 10, 11, 12, 13, 14, 15);
}

// A-role (row-major, K contiguous): lane = M row; 16 contig K
__device__ __forceinline__ HL loadA(const _Float16* __restrict__ Ah,
                                    const _Float16* __restrict__ Al,
                                    int lda, int mBase, int kBase, int lane) {
  int kofs = (lane < 16) ? 0 : 16;
  long o = (long)(mBase + (lane & 15)) * lda + kBase + kofs;
  HL r;
  r.h = ld16(Ah + o, Ah + o + 8);
  r.l = ld16(Al + o, Al + o + 8);
  return r;
}

// B-role (K-major / transposed storage BT[n*ldb+k]): lane = N col; 16 contig K
__device__ __forceinline__ HL loadB(const _Float16* __restrict__ Bh,
                                    const _Float16* __restrict__ Bl,
                                    int ldb, int kBase, int nBase, int lane) {
  int kofs = (lane < 16) ? 0 : 16;
  long o = (long)(nBase + (lane & 15)) * ldb + kBase + kofs;
  HL r;
  r.h = ld16(Bh + o, Bh + o + 8);
  r.l = ld16(Bl + o, Bl + o + 8);
  return r;
}

// ---------------------------------------------------------------------------
// Batched complex GEMM, pre-split f16 operands. Output: f32 planes and/or
// split-f16 planes (for chaining into the next GEMM without conversions).
// ---------------------------------------------------------------------------
__global__ void k_hgemm(const _Float16* __restrict__ Arh, const _Float16* __restrict__ Arl,
                        const _Float16* __restrict__ Aih, const _Float16* __restrict__ Ail,
                        int lda, long sA,
                        const _Float16* __restrict__ Brh, const _Float16* __restrict__ Brl,
                        const _Float16* __restrict__ Bih, const _Float16* __restrict__ Bil,
                        int ldb, long sB,
                        float* __restrict__ Cr, float* __restrict__ Ci, int ldc, long sCf,
                        _Float16* __restrict__ Hrh, _Float16* __restrict__ Hrl,
                        _Float16* __restrict__ Hih, _Float16* __restrict__ Hil, long sCh,
                        int Kk) {
  const int lane  = threadIdx.x;
  const int nBase = blockIdx.x * 16;
  const int mBase = blockIdx.y * 16;
  const int bz    = blockIdx.z;

  v8f P = {}, Q = {}, S = {}, T = {};
  for (int kb = 0; kb < Kk; kb += 32) {
    HL Ar = loadA(Arh + bz * sA, Arl + bz * sA, lda, mBase, kb, lane);
    HL Ai = loadA(Aih + bz * sA, Ail + bz * sA, lda, mBase, kb, lane);
    HL Br = loadB(Brh + bz * sB, Brl + bz * sB, ldb, kb, nBase, lane);
    HL Bi = loadB(Bih + bz * sB, Bil + bz * sB, ldb, kb, nBase, lane);
    P = mac3(Ar, Br, P);
    Q = mac3(Ai, Bi, Q);
    S = mac3(Ar, Bi, S);
    T = mac3(Ai, Br, T);
  }
  int n  = nBase + (lane & 15);
  int m0 = mBase + ((lane < 16) ? 0 : 8);
#pragma unroll
  for (int r = 0; r < 8; ++r) {
    float cre = P[r] - Q[r];
    float cim = S[r] + T[r];
    long o = (long)(m0 + r) * ldc + n;
    if (Cr) {
      Cr[bz * sCf + o] = cre;
      Ci[bz * sCf + o] = cim;
    }
    if (Hrh) {
      _Float16 h, l;
      split2(cre, h, l); Hrh[bz * sCh + o] = h; Hrl[bz * sCh + o] = l;
      split2(cim, h, l); Hih[bz * sCh + o] = h; Hil[bz * sCh + o] = l;
    }
  }
}

// ---------------------------------------------------------------------------
// rowT_z = W^T * (Ef .* H_z); H built on the fly. Two M-tiles per wave so the
// z-dependent B tile (sincos + split) is built once per 24 WMMAs.
// ---------------------------------------------------------------------------
__device__ __forceinline__ float dev_freq(int i) {
  int s = (i < NN / 2) ? i : i - NN;
  return (float)s * FREQ_SC;
}

__global__ void k_rowT(const _Float16* __restrict__ WTrh, const _Float16* __restrict__ WTrl,
                       const _Float16* __restrict__ WTih, const _Float16* __restrict__ WTil,
                       const float* __restrict__ Er, const float* __restrict__ Ei,
                       const float* __restrict__ zsv,
                       _Float16* __restrict__ Crh, _Float16* __restrict__ Crl,
                       _Float16* __restrict__ Cih, _Float16* __restrict__ Cil) {
  const int lane  = threadIdx.x;
  const int nBase = blockIdx.x * 16;   // v
  const int z     = blockIdx.y;
  const float zv  = zsv[z];
  const int l15   = lane & 15;
  const int bOfs  = (lane < 16) ? 0 : 16;
  const int n     = nBase + l15;
  const float fv  = dev_freq(n);
  const float fv2 = fv * fv;

  v8f P0 = {}, Q0 = {}, S0 = {}, T0 = {};
  v8f P1 = {}, Q1 = {}, S1 = {}, T1 = {};
  for (int kb = 0; kb < NP; kb += 32) {
    HL Ar0 = loadA(WTrh, WTrl, NP, 0,  kb, lane);
    HL Ai0 = loadA(WTih, WTil, NP, 0,  kb, lane);
    HL Ar1 = loadA(WTrh, WTrl, NP, 16, kb, lane);
    HL Ai1 = loadA(WTih, WTil, NP, 16, kb, lane);

    HL Br, Bi;
#pragma unroll
    for (int e = 0; e < 16; ++e) {     // B(k,n) = Ef[k,n] * H_z[k,n]
      int k = kb + bOfs + e;           // u
      float er = Er[(long)k * NP + n];
      float ei = Ei[(long)k * NP + n];
      float fu = dev_freq(k);
      float g  = sqrtf(fmaxf(1.0f - fu * fu - fv2, 0.0f));
      float sp, cp;
      __sincosf(KWAVE * g * zv, &sp, &cp);
      splitw(er * cp - ei * sp, Br.h, Br.l, e);
      splitw(er * sp + ei * cp, Bi.h, Bi.l, e);
    }
    P0 = mac3(Ar0, Br, P0);  Q0 = mac3(Ai0, Bi, Q0);
    S0 = mac3(Ar0, Bi, S0);  T0 = mac3(Ai0, Br, T0);
    P1 = mac3(Ar1, Br, P1);  Q1 = mac3(Ai1, Bi, Q1);
    S1 = mac3(Ar1, Bi, S1);  T1 = mac3(Ai1, Br, T1);
  }
  int mo = (lane < 16) ? 0 : 8;
  _Float16 h, l;
#pragma unroll
  for (int r = 0; r < 8; ++r) {
    long o0 = ((long)z * 32 + mo + r) * NP + n;
    long o1 = ((long)z * 32 + 16 + mo + r) * NP + n;
    split2(P0[r] - Q0[r], h, l); Crh[o0] = h; Crl[o0] = l;
    split2(S0[r] + T0[r], h, l); Cih[o0] = h; Cil[o0] = l;
    split2(P1[r] - Q1[r], h, l); Crh[o1] = h; Crl[o1] = l;
    split2(S1[r] + T1[r], h, l); Cih[o1] = h; Cil[o1] = l;
  }
}

// ---------------------------------------------------------------------------
// Constant tables (pre-split f16):
//   field = INCIDENT * cis(phase - c1)  -> stored transposed (B-role of GEMM1)
//   F     = DFT matrix / sqrt(N)        -> row-major (A-role; symmetric = B-role)
// ---------------------------------------------------------------------------
__global__ void k_init(const float* __restrict__ phase,
                       _Float16* __restrict__ fTrh, _Float16* __restrict__ fTrl,
                       _Float16* __restrict__ fTih, _Float16* __restrict__ fTil,
                       _Float16* __restrict__ Frh,  _Float16* __restrict__ Frl,
                       _Float16* __restrict__ Fih,  _Float16* __restrict__ Fil) {
  int idx = blockIdx.x * blockDim.x + threadIdx.x;
  if (idx >= NP * NP) return;
  int i = idx / NP, j = idx % NP;
  float fieldR = 0.f, fieldI = 0.f, FR = 0.f, FI = 0.f;
  if (i < NN && j < NN) {
    double x = (double)(j - NN / 2) * 1e-6;
    double y = (double)(i - NN / 2) * 1e-6;
    double r2 = x * x + y * y;
    double inc = exp(-r2 / (2.0 * 1.5e-4 * 1.5e-4));
    double c1  = fmod(D_PI / (5.32e-7 * 0.1) * r2, 2.0 * D_PI);
    double th  = (double)phase[i * NN + j] - c1;
    fieldR = (float)(inc * cos(th));
    fieldI = (float)(inc * sin(th));
    int t = (int)(((long)i * (long)j) % NN);           // exact twiddle reduction
    double a = -2.0 * D_PI * (double)t / (double)NN;
    const double sc = 0.044721359549995794;            // 1/sqrt(500) conditioning
    FR = (float)(cos(a) * sc);
    FI = (float)(sin(a) * sc);
  }
  _Float16 h, l;
  long tidx = (long)j * NP + i;                        // transposed (n=j, k=i)
  split2(fieldR, h, l); fTrh[tidx] = h; fTrl[tidx] = l;
  split2(fieldI, h, l); fTih[tidx] = h; fTil[tidx] = l;
  split2(FR, h, l); Frh[idx] = h; Frl[idx] = l;
  split2(FI, h, l); Fih[idx] = h; Fil[idx] = l;
}

// WT[m,k] = W[k,m] = cis(+2*pi*k*(235+m)/500); serves A-role of k_rowT (row-
// major over m) AND B-role of the crop GEMM (K-major over k). 32x512.
__global__ void k_init_w(_Float16* __restrict__ WTrh, _Float16* __restrict__ WTrl,
                         _Float16* __restrict__ WTih, _Float16* __restrict__ WTil) {
  int idx = blockIdx.x * blockDim.x + threadIdx.x;
  if (idx >= 32 * NP) return;
  int m = idx / NP, k = idx % NP;
  float wr = 0.f, wi = 0.f;
  if (k < NN && m < PP) {
    int t = (int)(((long)k * (long)(NN / 2 - RRAD + m)) % NN);
    double a = 2.0 * D_PI * (double)t / (double)NN;
    wr = (float)cos(a);
    wi = (float)sin(a);
  }
  _Float16 h, l;
  split2(wr, h, l); WTrh[idx] = h; WTrl[idx] = l;
  split2(wi, h, l); WTih[idx] = h; WTil[idx] = l;
}

// |crop|^2, normalize to PHOTONS per z
__global__ void k_intensity(const float* __restrict__ cr, const float* __restrict__ ci,
                            float* __restrict__ psf) {
  int z = blockIdx.x, tid = threadIdx.x;
  __shared__ float red[256];
  const float* zr = cr + (long)z * 32 * 32;
  const float* zi = ci + (long)z * 32 * 32;
  float acc = 0.f;
  for (int p = tid; p < PP * PP; p += 256) {
    int dy = p / PP, dx = p % PP;
    float re = zr[dy * 32 + dx], im = zi[dy * 32 + dx];
    float v = re * re + im * im;
    psf[(long)z * PP * PP + p] = v;
    acc += v;
  }
  red[tid] = acc; __syncthreads();
  for (int s = 128; s > 0; s >>= 1) {
    if (tid < s) red[tid] += red[tid + s];
    __syncthreads();
  }
  float scale = 1.0e6f / (red[0] + 1e-12f);
  for (int p = tid; p < PP * PP; p += 256) psf[(long)z * PP * PP + p] *= scale;
}

__global__ void k_zero(float* __restrict__ p, int n) {
  int idx = blockIdx.x * blockDim.x + threadIdx.x;
  if (idx < n) p[idx] = 0.f;
}

// scatter-add 31x31 psf tiles at xyz positions (emitters overlap -> atomics)
__global__ void k_scatter(const float* __restrict__ psf, const int* __restrict__ xyz,
                          float* __restrict__ canvas) {
  int b = blockIdx.x >> 6, e = blockIdx.x & 63;
  int r0 = xyz[((long)b * NE + e) * 2 + 0] - RRAD;
  int c0 = xyz[((long)b * NE + e) * 2 + 1] - RRAD;
  float* cb = canvas + (long)b * CANV * CANV;
  const float* pe = psf + (long)e * PP * PP;
  for (int p = threadIdx.x; p < PP * PP; p += blockDim.x) {
    int dy = p / PP, dx = p % PP;
    atomicAdd(&cb[(r0 + dy) * CANV + (c0 + dx)], pe[p]);
  }
}

// fused 7x7 blur (pad 2 -> 198x198) + camera noise chain
__global__ void k_blur_noise(const float* __restrict__ canvas,
                             const float* __restrict__ std_u,
                             const float* __restrict__ eps_dark,
                             const float* __restrict__ eps_photon,
                             const float* __restrict__ eps_read,
                             float* __restrict__ out, int total) {
  __shared__ float kk[49];
  int tid = threadIdx.x;
  if (tid < 49) {
    float stdv = 0.8f + 0.4f * std_u[0];
    float s2 = stdv * stdv;
    int u = tid / 7, v = tid % 7;
    float d2 = (float)((u - 3) * (u - 3) + (v - 3) * (v - 3));
    kk[tid] = __expf(-d2 / (2.0f * s2)) / (2.0f * (float)D_PI * s2);
  }
  __syncthreads();
  int idx = blockIdx.x * blockDim.x + tid;
  if (idx >= total) return;
  int b = idx / (OUTW * OUTW);
  int rem = idx - b * OUTW * OUTW;
  int i = rem / OUTW, j = rem % OUTW;
  const float* cb = canvas + (long)b * CANV * CANV;
  float acc = 0.f;
#pragma unroll
  for (int u = 0; u < 7; ++u) {
    int ri = i + u - 2;
    if (ri < 0 || ri >= CANV) continue;
#pragma unroll
    for (int v = 0; v < 7; ++v) {
      int cj = j + v - 2;
      if (cj < 0 || cj >= CANV) continue;
      acc += kk[u * 7 + v] * cb[ri * CANV + cj];
    }
  }
  float sig   = acc * 0.9f;                                   // QE
  float dark  = 0.005f + eps_dark[idx] * sqrtf(0.005f);
  float tot   = fmaxf(sig + dark, 0.0f);
  float noisy = tot + eps_photon[idx] * sqrtf(fmaxf(tot, 1e-12f));
  float elec  = noisy + eps_read[idx] * 1.6f;
  float adu   = fminf(fmaxf(elec * 2.0f, 0.0f), 65535.0f);
  float o     = (adu <= 10.0f) ? 1.0f : fminf(adu, 4.0e9f);
  out[idx] = o / 4.0e9f;
}

// ---------------------------------------------------------------------------
extern "C" void kernel_launch(void* const* d_in, const int* in_sizes, int n_in,
                              void* d_out, int out_size, void* d_ws, size_t ws_size,
                              hipStream_t stream) {
  const float* phase      = (const float*)d_in[0];
  const float* zsv        = (const float*)d_in[1];
  const int*   xyz        = (const int*)d_in[2];
  const float* std_u      = (const float*)d_in[3];
  const float* eps_dark   = (const float*)d_in[4];
  const float* eps_photon = (const float*)d_in[5];
  const float* eps_read   = (const float*)d_in[6];
  float* out = (float*)d_out;

  char* p = (char*)d_ws;
  auto allocH = [&](size_t n) { _Float16* r = (_Float16*)p; p += n * sizeof(_Float16); return r; };
  auto allocF = [&](size_t n) { float* r = (float*)p; p += n * sizeof(float); return r; };

  const size_t SZ = (size_t)NP * NP;
  _Float16 *Frh  = allocH(SZ), *Frl  = allocH(SZ), *Fih  = allocH(SZ), *Fil  = allocH(SZ);
  _Float16 *fTrh = allocH(SZ), *fTrl = allocH(SZ), *fTih = allocH(SZ), *fTil = allocH(SZ);
  _Float16 *Grh  = allocH(SZ), *Grl  = allocH(SZ), *Gih  = allocH(SZ), *Gil  = allocH(SZ);
  float    *EfRe = allocF(SZ), *EfIm = allocF(SZ);
  _Float16 *WTrh = allocH(32 * NP), *WTrl = allocH(32 * NP);
  _Float16 *WTih = allocH(32 * NP), *WTil = allocH(32 * NP);
  const size_t RT = (size_t)NE * 32 * NP;
  _Float16 *rTrh = allocH(RT), *rTrl = allocH(RT), *rTih = allocH(RT), *rTil = allocH(RT);
  float *cropRe = allocF((size_t)NE * 32 * 32), *cropIm = allocF((size_t)NE * 32 * 32);
  float *psf    = allocF((size_t)NE * PP * PP);
  float *canvas = allocF((size_t)NB * CANV * CANV);

  k_init<<<(NP * NP + 255) / 256, 256, 0, stream>>>(
      phase, fTrh, fTrl, fTih, fTil, Frh, Frl, Fih, Fil);
  k_init_w<<<(32 * NP + 255) / 256, 256, 0, stream>>>(WTrh, WTrl, WTih, WTil);

  // G = F * field  -> split-f16 out (A-role of next GEMM)
  k_hgemm<<<dim3(NP / 16, NP / 16, 1), 32, 0, stream>>>(
      Frh, Frl, Fih, Fil, NP, 0,
      fTrh, fTrl, fTih, fTil, NP, 0,
      nullptr, nullptr, NP, 0,
      Grh, Grl, Gih, Gil, 0, NP);
  // Ef = G * F  -> f32 out (consumed by on-the-fly H multiply)
  k_hgemm<<<dim3(NP / 16, NP / 16, 1), 32, 0, stream>>>(
      Grh, Grl, Gih, Gil, NP, 0,
      Frh, Frl, Fih, Fil, NP, 0,
      EfRe, EfIm, NP, 0,
      nullptr, nullptr, nullptr, nullptr, 0, NP);

  // rowT_z = W^T * (Ef .* H_z)  -> split-f16 out, all 64 z
  k_rowT<<<dim3(NP / 16, NE), 32, 0, stream>>>(
      WTrh, WTrl, WTih, WTil, EfRe, EfIm, zsv, rTrh, rTrl, rTih, rTil);

  // crop_z = rowT_z * W  -> f32 out (batched; WT reused as K-major B operand)
  k_hgemm<<<dim3(2, 2, NE), 32, 0, stream>>>(
      rTrh, rTrl, rTih, rTil, NP, (long)32 * NP,
      WTrh, WTrl, WTih, WTil, NP, 0,
      cropRe, cropIm, 32, (long)32 * 32,
      nullptr, nullptr, nullptr, nullptr, 0, NP);

  k_intensity<<<NE, 256, 0, stream>>>(cropRe, cropIm, psf);
  k_zero<<<(NB * CANV * CANV + 255) / 256, 256, 0, stream>>>(canvas, NB * CANV * CANV);
  k_scatter<<<NB * NE, 256, 0, stream>>>(psf, xyz, canvas);
  k_blur_noise<<<(NB * OUTW * OUTW + 255) / 256, 256, 0, stream>>>(
      canvas, std_u, eps_dark, eps_photon, eps_read, out, NB * OUTW * OUTW);
}
